// FFTGADBase_85478439125950
// MI455X (gfx1250) — compile-verified, matrix operand order checked by reading, not run on hardware
//
#include <hip/hip_runtime.h>
#include <hip/hip_bf16.h>

#define L_C 0.24f
#define K_C 0.03f
#define EPS_C 1e-8f
#define THRESH_C 0.1f
#define B_ 2
#define H_ 512
#define W_ 512
#define SH_ 64
#define SW_ 64
#define NPRE_ 256
#define FFT_STEPS_ 10
#define STEP_ 48
#define NB_ 11

typedef __attribute__((ext_vector_type(16))) _Float16 v16h;
typedef __attribute__((ext_vector_type(8)))  float    v8f;
typedef __attribute__((ext_vector_type(4)))  unsigned int u32x4;
typedef __attribute__((ext_vector_type(8)))  int i32x8;
typedef __attribute__((ext_vector_type(4)))  int i32x4;

#if defined(__HIP_DEVICE_COMPILE__) && defined(__gfx1250__) && __has_builtin(__builtin_amdgcn_tensor_load_to_lds)
#define USE_TDM 1
#else
#define USE_TDM 0
#endif

__device__ __forceinline__ float gfun(float m) {
    return 1.0f / (1.0f + (m * m) * (1.0f / (K_C * K_C)));
}

// ---------------- Kernel 1: edge-stopping coefficients cv, ch ----------------
__global__ __launch_bounds__(256) void coeffs_kernel(const float* __restrict__ guide,
                                                     const float* __restrict__ img,
                                                     float* __restrict__ cv,
                                                     float* __restrict__ ch) {
    int idx = blockIdx.x * 256 + threadIdx.x;
    const int HW = H_ * W_;
    if (idx < B_ * (H_ - 1) * W_) {
        int x = idx % W_;
        int y = (idx / W_) % (H_ - 1);
        int b = idx / (W_ * (H_ - 1));
        const float* g = guide + (size_t)b * 3 * HW;
        const float* im = img + (size_t)b * HW;
        int p0 = y * W_ + x, p1 = p0 + W_;
        float m = fabsf(g[p1] - g[p0]) + fabsf(g[HW + p1] - g[HW + p0]) +
                  fabsf(g[2 * HW + p1] - g[2 * HW + p0]) + fabsf(im[p1] - im[p0]);
        cv[idx] = gfun(0.25f * m);
    }
    if (idx < B_ * H_ * (W_ - 1)) {
        int x = idx % (W_ - 1);
        int y = (idx / (W_ - 1)) % H_;
        int b = idx / ((W_ - 1) * H_);
        const float* g = guide + (size_t)b * 3 * HW;
        const float* im = img + (size_t)b * HW;
        int p0 = y * W_ + x, p1 = p0 + 1;
        float m = fabsf(g[p1] - g[p0]) + fabsf(g[HW + p1] - g[HW + p0]) +
                  fabsf(g[2 * HW + p1] - g[2 * HW + p0]) + fabsf(im[p1] - im[p0]);
        ch[idx] = gfun(0.25f * m);
    }
}

// ---------------- Kernel 2: uniform regions (3x3 box variance) ----------------
__global__ __launch_bounds__(256) void uniform_kernel(const float* __restrict__ cv,
                                                      const float* __restrict__ ch,
                                                      float* __restrict__ u) {
    int idx = blockIdx.x * 256 + threadIdx.x;
    const int UW = W_ - 1, UH = H_ - 1;
    if (idx >= B_ * UH * UW) return;
    int x = idx % UW;
    int y = (idx / UW) % UH;
    int b = idx / (UW * UH);
    const float* CV = cv + (size_t)b * (H_ - 1) * W_;
    const float* CH = ch + (size_t)b * H_ * (W_ - 1);
    float s1 = 0.f, s2 = 0.f, t1 = 0.f, t2 = 0.f;
    for (int dy = -1; dy <= 1; ++dy)
        for (int dx = -1; dx <= 1; ++dx) {
            int yy = y + dy, xx = x + dx;
            if (yy >= 0 && yy < (H_ - 1) && xx >= 0 && xx < W_) {
                float v = CV[yy * W_ + xx]; s1 += v; s2 += v * v;
            }
            if (yy >= 0 && yy < H_ && xx >= 0 && xx < (W_ - 1)) {
                float v = CH[yy * (W_ - 1) + xx]; t1 += v; t2 += v * v;
            }
        }
    float m1 = s1 * (1.f / 9.f); float varv = s2 * (1.f / 9.f) - m1 * m1;
    float m2 = t1 * (1.f / 9.f); float varh = t2 * (1.f / 9.f) - m2 * m2;
    u[idx] = (varv < THRESH_C && varh < THRESH_C) ? 1.0f : 0.0f;
}

// ---------------- Kernel 3: block FFT diffusion ----------------
// rfft2 * op^10 * irfft2 == 10 explicit periodic Jacobi steps with constant
// coefficients (op is exactly the eigenvalue of one periodic 5-point step).
__device__ float block_reduce_sum(float v, float* red) {
    int tid = threadIdx.x;
    red[tid] = v;
    __syncthreads();
    for (int off = 128; off > 0; off >>= 1) {
        if (tid < off) red[tid] += red[tid + off];
        __syncthreads();
    }
    float r = red[0];
    __syncthreads();
    return r;
}

__global__ __launch_bounds__(256) void fftblock_kernel(const float* __restrict__ img,
                                                       const float* __restrict__ cv,
                                                       const float* __restrict__ ch,
                                                       const float* __restrict__ u,
                                                       float* __restrict__ out) {
    __shared__ float A[64][65];
    __shared__ float Bb[64][65];
    __shared__ float red[256];
    int bid = blockIdx.x;
    int ixb = bid % NB_;
    int iyb = (bid / NB_) % NB_;
    int b = bid / (NB_ * NB_);
    int y0 = iyb * STEP_, x0 = ixb * STEP_;
    int y1 = y0 + 64; if (y1 > H_) y1 = H_;
    int x1 = x0 + 64; if (x1 > W_) x1 = W_;
    int h = y1 - y0, w = x1 - x0, hw = h * w;
    int tid = threadIdx.x;
    const float* I = img + (size_t)b * H_ * W_;

    for (int i = tid; i < hw; i += 256) {
        int r = i / w, c = i % w;
        A[r][c] = I[(y0 + r) * W_ + x0 + c];
    }
    __syncthreads();

    const float* CV = cv + (size_t)b * (H_ - 1) * W_;
    int ylv = y1 - 1; if (ylv > H_ - 1) ylv = H_ - 1;
    int n1 = (ylv - y0) * w;
    float s = 0.f;
    for (int i = tid; i < n1; i += 256) { int r = i / w, c = i % w; s += CV[(y0 + r) * W_ + x0 + c]; }
    float cvm = block_reduce_sum(s, red) / (float)n1;

    const float* CH = ch + (size_t)b * H_ * (W_ - 1);
    int xlh = x1 - 1; if (xlh > W_ - 1) xlh = W_ - 1;
    int wh = xlh - x0, n2 = h * wh;
    s = 0.f;
    for (int i = tid; i < n2; i += 256) { int r = i / wh, c = i % wh; s += CH[(y0 + r) * (W_ - 1) + x0 + c]; }
    float chm = block_reduce_sum(s, red) / (float)n2;

    const float* U = u + (size_t)b * (H_ - 1) * (W_ - 1);
    int yu = y1; if (yu > H_ - 1) yu = H_ - 1;
    int xu = x1; if (xu > W_ - 1) xu = W_ - 1;
    int wu = xu - x0, n3 = (yu - y0) * wu;
    s = 0.f;
    for (int i = tid; i < n3; i += 256) { int r = i / wu, c = i % wu; s += U[(y0 + r) * (W_ - 1) + x0 + c]; }
    float frac = block_reduce_sum(s, red) / (float)n3;

    float av = L_C * cvm, ah = L_C * chm;
    float (*S)[65] = A; float (*D)[65] = Bb;
    for (int it = 0; it < FFT_STEPS_; ++it) {
        for (int i = tid; i < hw; i += 256) {
            int r = i / w, c = i % w;
            float x = S[r][c];
            int rm = (r == 0) ? (h - 1) : (r - 1), rp = (r == h - 1) ? 0 : (r + 1);
            int cm = (c == 0) ? (w - 1) : (c - 1), cp = (c == w - 1) ? 0 : (c + 1);
            D[r][c] = x + av * (S[rp][c] + S[rm][c] - 2.f * x) + ah * (S[r][cp] + S[r][cm] - 2.f * x);
        }
        __syncthreads();
        float (*t)[65] = S; S = D; D = t;
    }

    int oy1 = y0 + STEP_; if (oy1 > y1) oy1 = y1;
    int ox1 = x0 + STEP_; if (ox1 > x1) ox1 = x1;
    int ow = ox1 - x0, on = (oy1 - y0) * ow;
    bool blendOn = (y0 > 0) || (x0 > 0);
    bool useNew = frac > 0.7f;
    float* O = out + (size_t)b * H_ * W_;
    for (int i = tid; i < on; i += 256) {
        int r = i / ow, c = i % ow;
        float cur = I[(y0 + r) * W_ + x0 + c];
        float res;
        if (useNew) {
            float nv = S[r][c];
            if (blendOn) {
                float ry = (y0 > 0) ? ((r < 16) ? (float)r * (1.f / 16.f) : 1.f) : 1.f;
                float rx = (x0 > 0) ? ((c < 16) ? (float)c * (1.f / 16.f) : 1.f) : 1.f;
                float bl = ry * rx;
                res = cur * (1.f - bl) + nv * bl;
            } else res = nv;
        } else res = cur;
        O[(y0 + r) * W_ + x0 + c] = res;
    }
}

// ---------------- Kernel 4: fused diffuse + 8x8 ratio adjust (x256) ----------------
// 32x32 tile per workgroup (8 wave32). Core staged by TDM tensor_load_to_lds with
// hardware pad insertion (pitch 36). Branchless edge handling (clamped loads +
// cndmask-zeroed weights). 8x8 cell sums via v_wmma_f32_16x16x32_f16.
__global__ __launch_bounds__(256) void iter_kernel(const float* __restrict__ in,
                                                   float* __restrict__ out,
                                                   const float* __restrict__ cv,
                                                   const float* __restrict__ ch,
                                                   const float* __restrict__ src,
                                                   const float* __restrict__ msk) {
    __shared__ float T[34][36];      // halo tile, pitch 36 = 32 data + 4 TDM pad DWORDs
    __shared__ float O[32][33];      // diffused values
    __shared__ _Float16 P[16][16];   // partial cell sums, P[cell][sub]
    __shared__ float R[16];          // per-cell ratios

    int tx0 = blockIdx.x * 32, ty0 = blockIdx.y * 32, b = blockIdx.z;
    const float* I = in + (size_t)b * H_ * W_;
    int tid = threadIdx.x;
    int wv = tid >> 5, ln = tid & 31;

#if USE_TDM
    if (wv == 0) {
        unsigned ldsoff = (unsigned)(unsigned long long)(&T[1][1]);
        unsigned long long ga = (unsigned long long)(const void*)(I + ty0 * W_ + tx0);
        u32x4 g0 = { 1u,                                   // count=1, gather off
                     ldsoff,                               // lds_addr
                     (unsigned)(ga & 0xffffffffu),         // global_addr lo
                     (unsigned)((ga >> 32) & 0x01ffffffu) | (2u << 30) }; // addr hi + type=2
        i32x8 g1 = { (int)((2u << 16) | (1u << 20) | (4u << 22) | (3u << 25)),
                     // data_size=4B, pad_enable, pad_interval=32 DW, pad_amount=4 DW
                     (int)(512u << 16),                    // tensor_dim0 = 512
                     (int)(512u << 16),                    // tensor_dim1 = 512
                     (int)(32u << 16),                     // tile_dim0 = 32
                     32,                                   // tile_dim1 = 32, tile_dim2 = 0
                     512,                                  // tensor_dim0_stride = 512
                     0, 0 };
        i32x4 z4 = { 0, 0, 0, 0 };
#if __clang_major__ >= 23
        i32x8 z8 = { 0, 0, 0, 0, 0, 0, 0, 0 };
        __builtin_amdgcn_tensor_load_to_lds(g0, g1, z4, z4, z8, 0);
#else
        __builtin_amdgcn_tensor_load_to_lds(g0, g1, z4, z4, 0);
#endif
    }
#else
    for (int i = tid; i < 32 * 32; i += 256) {
        int r = i >> 5, c = i & 31;
        T[1 + r][1 + c] = I[(ty0 + r) * W_ + tx0 + c];
    }
#endif

    // halo ring (132 elements), zero outside image (weights are zeroed there)
    if (tid < 132) {
        int r, c;
        if (tid < 34)       { r = 0;  c = tid; }
        else if (tid < 68)  { r = 33; c = tid - 34; }
        else if (tid < 100) { r = tid - 68 + 1;  c = 0; }
        else                { r = tid - 100 + 1; c = 33; }
        int gy = ty0 + r - 1, gx = tx0 + c - 1;
        float v = 0.f;
        if (gy >= 0 && gy < H_ && gx >= 0 && gx < W_) v = I[gy * W_ + gx];
        T[r][c] = v;
    }

#if USE_TDM
    if (wv == 0) __builtin_amdgcn_s_wait_tensorcnt(0);
#endif
    __syncthreads();

    // each thread owns 4 rows x 1 col; all 4 pixels lie in one 8x8 cell
    int c = ln;
    int r0 = wv * 4;
    int gx = tx0 + c;
    int gy0 = ty0 + r0;
    const float* CV = cv + (size_t)b * (H_ - 1) * W_;
    const float* CH = ch + (size_t)b * H_ * (W_ - 1);

    // vertical weights: north of row r == south of row r-1 -> 5 loads, clamped
    float cvw[5];
#pragma unroll
    for (int k = 0; k < 5; ++k) {
        int yy = gy0 - 1 + k;
        int yc = (yy < 0) ? 0 : ((yy > (H_ - 2)) ? (H_ - 2) : yy);
        cvw[k] = L_C * CV[yc * W_ + gx];
    }
    cvw[0] = (gy0 == 0)      ? 0.f : cvw[0];
    cvw[4] = (gy0 + 4 == H_) ? 0.f : cvw[4];

    int xw = (gx == 0) ? 0 : (gx - 1);
    int xe = (gx >= W_ - 1) ? (W_ - 2) : gx;
    bool hasW = (gx >= 1), hasE = (gx <= W_ - 2);

    float p = 0.f;
#pragma unroll
    for (int k = 0; k < 4; ++k) {
        int r = r0 + k;
        int gy = gy0 + k;
        float x = T[1 + r][1 + c];
        float kE = L_C * CH[gy * (W_ - 1) + xe]; kE = hasE ? kE : 0.f;
        float kW = L_C * CH[gy * (W_ - 1) + xw]; kW = hasW ? kW : 0.f;
        float nv = x
                 + cvw[k + 1] * (T[2 + r][1 + c] - x)
                 - cvw[k]     * (x - T[r][1 + c])
                 + kE * (T[1 + r][2 + c] - x)
                 - kW * (x - T[1 + r][c]);
        O[r][c] = nv;
        p += nv;
    }
    int cell = (wv >> 1) * 4 + (c >> 3);
    int sub  = (wv & 1) * 8 + (c & 7);
    P[cell][sub] = (_Float16)p;
    __syncthreads();

    // wave 0: cell sums via WMMA (A = partials, B = ones -> D[m,*] = cellsum[m])
    if (wv == 0) {
        int row = ln & 15, half = ln >> 4;
        v16h a;
#pragma unroll
        for (int j = 0; j < 8; ++j) a[j] = P[row][half * 8 + j];
#pragma unroll
        for (int j = 8; j < 16; ++j) a[j] = (_Float16)0.f;
        v16h bb;
#pragma unroll
        for (int j = 0; j < 16; ++j) bb[j] = (_Float16)1.0f;
        v8f acc = {};
        acc = __builtin_amdgcn_wmma_f32_16x16x32_f16(false, a, false, bb,
                                                     (short)0, acc, false, false);
        if (ln == 0 || ln == 16) {
            int base = (ln == 0) ? 0 : 8;
            const float* S = src + (size_t)b * SH_ * SW_;
            const float* M = msk + (size_t)b * SH_ * SW_;
            int cy0 = ty0 >> 3, cx0 = tx0 >> 3;
#pragma unroll
            for (int j = 0; j < 8; ++j) {
                int cm = base + j;
                int cy = cy0 + (cm >> 2), cx = cx0 + (cm & 3);
                float dsum = acc[j] * (1.0f / 64.0f);
                float mv = M[cy * SW_ + cx];
                float sv = S[cy * SW_ + cx];
                R[cm] = (mv < 0.5f) ? 1.0f : (sv / (dsum + EPS_C));
            }
        }
    }
    __syncthreads();

    float* Ob = out + (size_t)b * H_ * W_;
    float ra = R[cell];
#pragma unroll
    for (int k = 0; k < 4; ++k) {
        int r = r0 + k;
        Ob[(ty0 + r) * W_ + gx] = O[r][c] * ra;
    }
}

// ---------------- Host launch ----------------
extern "C" void kernel_launch(void* const* d_in, const int* in_sizes, int n_in,
                              void* d_out, int out_size, void* d_ws, size_t ws_size,
                              hipStream_t stream) {
    (void)in_sizes; (void)n_in; (void)out_size; (void)ws_size;
    const float* guide  = (const float*)d_in[0];
    const float* source = (const float*)d_in[1];
    const float* mask   = (const float*)d_in[2];
    const float* ybic   = (const float*)d_in[3];

    float* ws = (float*)d_ws;
    const size_t cvN = (size_t)B_ * (H_ - 1) * W_;
    const size_t chN = (size_t)B_ * H_ * (W_ - 1);
    const size_t uN  = (size_t)B_ * (H_ - 1) * (W_ - 1);
    const size_t iN  = (size_t)B_ * H_ * W_;
    float* cv   = ws; ws += cvN;
    float* ch   = ws; ws += chN;
    float* u    = ws; ws += uN;
    float* imgA = ws; ws += iN;
    float* imgB = ws; ws += iN;

    int nmax = (int)((cvN > chN) ? cvN : chN);
    coeffs_kernel<<<(nmax + 255) / 256, 256, 0, stream>>>(guide, ybic, cv, ch);
    uniform_kernel<<<((int)uN + 255) / 256, 256, 0, stream>>>(cv, ch, u);
    fftblock_kernel<<<B_ * NB_ * NB_, 256, 0, stream>>>(ybic, cv, ch, u, imgA);

    dim3 gD(W_ / 32, H_ / 32, B_);
    const float* pin = imgA;
    float* pout = imgB;
    for (int i = 0; i < NPRE_; ++i) {
        float* o = (i == NPRE_ - 1) ? (float*)d_out : pout;
        iter_kernel<<<gD, 256, 0, stream>>>(pin, o, cv, ch, source, mask);
        pin = o;
        pout = (o == imgB) ? imgA : imgB;
    }
}
